// PLE_57861799411839
// MI455X (gfx1250) — matrix operand order repaired and checked by reading, not compiled, not run on hardware
//
#include <hip/hip_runtime.h>
#include <hip/hip_bf16.h>

typedef __attribute__((ext_vector_type(16))) _Float16 v16h;
typedef __attribute__((ext_vector_type(8)))  _Float16 v8h;
typedef __attribute__((ext_vector_type(2)))  _Float16 v2h;
typedef __attribute__((ext_vector_type(8)))  float    v8f;

#define TOKENS   65536      // B*S
#define H        256
#define HE       512        // H*E
#define XP       272        // LDS row pitch (halves), 256 + 16 pad
#define WTSZ     131072     // 512*256 halves per expert matrix

struct PleArgs {
    const float *xsh, *xt1, *xt2;                 // input streams [65536,256]
    const float *bsh, *bt1, *bt2;                 // expert biases [2,512]
    const float *Wgsh, *bgsh, *Wgt1, *bgt1, *Wgt2, *bgt2;
    const float *Wspan, *bspan, *Was, *bas, *Wae, *bae;
    const _Float16 *wt;                           // f16 W^T [l][e][512][256]
    float *out;
};

// ---------- prep: convert + transpose expert weights to f16 ----------
__global__ __launch_bounds__(256) void prep_weights(const float* __restrict__ Wsh,
                                                    const float* __restrict__ Wt1,
                                                    const float* __restrict__ Wt2,
                                                    _Float16* __restrict__ wt) {
    int i = blockIdx.x * 256 + threadIdx.x;       // 786432 total
    if (i >= 2 * 3 * WTSZ) return;
    int k  = i & 255;             // K (input dim), contiguous in dst
    int n  = (i >> 8) & 511;      // N (expert col)
    int le = i >> 17;             // 0..5  (= l*3 + e)
    int l  = le / 3, e = le % 3;
    const float* W = (e == 0) ? Wsh : (e == 1) ? Wt1 : Wt2;   // [l][k][n] row-major
    wt[i] = (_Float16)W[l * (H * HE) + k * HE + n];
}

// ---------- fused PLE: both layers + heads ----------
__global__ __launch_bounds__(256) void ple_fused(PleArgs p) {
    __shared__ __align__(16) _Float16 sX[3][16 * XP];   // sh/t1/t2 token tiles (f16)
    __shared__ __align__(16) _Float16 sScr[8 * 1536];   // per-wave 16x32x3 expert scratch (also gw staging)
    __shared__ float sGate[16][16];                     // 14 gate cols used, pitch 16

    const int tid  = threadIdx.x;
    const int lane = tid & 31;
    const int wave = tid >> 5;
    const int m0   = blockIdx.x * 16;

    // load X tiles, convert f32 -> f16
    for (int s = 0; s < 3; ++s) {
        const float* src = (s == 0) ? p.xsh : (s == 1) ? p.xt1 : p.xt2;
        for (int i = tid; i < 16 * H; i += 256) {
            int r = i >> 8, k = i & 255;
            sX[s][r * XP + k] = (_Float16)src[(size_t)(m0 + r) * H + k];
        }
    }
    __syncthreads();

    for (int l = 0; l < 2; ++l) {
        // ---- stage gate weights [k][c] (c: 0-5 gsh, 6-9 g1, 10-13 g2) into scratch as f32 ----
        float* gw = reinterpret_cast<float*>(sScr);
        for (int i = tid; i < 256 * 14; i += 256) {
            int k = i / 14, c = i % 14;
            float w;
            if (c < 6)       w = p.Wgsh[l * (H * 6) + k * 6 + c];
            else if (c < 10) w = p.Wgt1[l * (H * 4) + k * 4 + (c - 6)];
            else             w = p.Wgt2[l * (H * 4) + k * 4 + (c - 10)];
            gw[i] = w;
        }
        __syncthreads();

        // ---- gate GEMVs: 16 tokens x 14 cols ----
        if (tid < 224) {
            int tok = tid / 14, c = tid % 14;
            int s = (c < 6) ? 0 : (c < 10) ? 1 : 2;
            const _Float16* xr = &sX[s][tok * XP];
            float acc = 0.f;
            for (int k = 0; k < H; ++k) acc += (float)xr[k] * gw[k * 14 + c];
            float bias = (c < 6) ? p.bgsh[l * 6 + c]
                       : (c < 10) ? p.bgt1[l * 4 + (c - 6)]
                                  : p.bgt2[l * 4 + (c - 10)];
            sGate[tok][c] = acc + bias;
        }
        __syncthreads();

        // ---- softmax per token per gate group ----
        if (tid < 48) {
            int grp = tid >> 4, tok = tid & 15;
            int c0 = (grp == 0) ? 0 : (grp == 1) ? 6 : 10;
            int n  = (grp == 0) ? 6 : 4;
            float mx = -1e30f;
            for (int c = 0; c < n; ++c) mx = fmaxf(mx, sGate[tok][c0 + c]);
            float ss = 0.f, ev[6];
            for (int c = 0; c < n; ++c) { ev[c] = __expf(sGate[tok][c0 + c] - mx); ss += ev[c]; }
            float inv = 1.f / ss;
            for (int c = 0; c < n; ++c) sGate[tok][c0 + c] = ev[c] * inv;
        }
        __syncthreads();

        // ---- expert GEMMs (WMMA) + gated combine; wave owns 2 output N-tiles of 16 h ----
        v8f res[2][3] = {};
        _Float16* scr = &sScr[wave * 1536];
        const _Float16* wtL = p.wt + (size_t)l * 3 * WTSZ;
        const int row  = lane & 15;
        const int koff = (lane >> 4) * 16;    // A-frag K sub-offset per lane group

        for (int t = 0; t < 2; ++t) {
            const int h0 = (wave * 2 + t) * 16;   // output h base
            const int c0 = 2 * h0;                // expert full-col base (32 wide)

            v8f acc[3][2] = {};
            for (int ks = 0; ks < 8; ++ks) {
                const int kb = ks * 32;
                v16h a[3];
                #pragma unroll
                for (int s = 0; s < 3; ++s) {
                    const _Float16* xr = &sX[s][row * XP + kb + koff];
                    v8h lo = *(const v8h*)xr;
                    v8h hi = *(const v8h*)(xr + 8);
                    a[s] = __builtin_shufflevector(lo, hi, 0,1,2,3,4,5,6,7,8,9,10,11,12,13,14,15);
                }
                #pragma unroll
                for (int e = 0; e < 3; ++e) {
                    #pragma unroll
                    for (int sub = 0; sub < 2; ++sub) {
                        const int col = c0 + sub * 16 + (lane & 15);
                        const _Float16* bp = wtL + e * WTSZ + col * H + kb + (lane >> 4) * 16;
                        v16h b = *(const v16h*)bp;
                        acc[e][sub] = __builtin_amdgcn_wmma_f32_16x16x32_f16(
                            false, a[e], false, b, (short)0, acc[e][sub], false, false);
                    }
                }
            }

            // bias + relu + stage to wave-private scratch (f16, [e][M=16][j=32])
            #pragma unroll
            for (int e = 0; e < 3; ++e) {
                const float* bptr = (e == 0) ? p.bsh : (e == 1) ? p.bt1 : p.bt2;
                #pragma unroll
                for (int sub = 0; sub < 2; ++sub) {
                    float bias = bptr[l * HE + c0 + sub * 16 + (lane & 15)];
                    #pragma unroll
                    for (int r = 0; r < 8; ++r) {
                        float v = fmaxf(acc[e][sub][r] + bias, 0.f);
                        int M = (lane < 16) ? r : r + 8;
                        scr[e * 512 + M * 32 + sub * 16 + (lane & 15)] = (_Float16)v;
                    }
                }
            }

            // gated combine: result element (M, h0 + (lane&15)) in reg r (C-matrix layout)
            const int hl = lane & 15;
            #pragma unroll
            for (int r = 0; r < 8; ++r) {
                int M = (lane < 16) ? r : r + 8;
                const float* g = &sGate[M][0];
                v2h pes = *(const v2h*)&scr[0 * 512 + M * 32 + 2 * hl];
                v2h pe1 = *(const v2h*)&scr[1 * 512 + M * 32 + 2 * hl];
                v2h pe2 = *(const v2h*)&scr[2 * 512 + M * 32 + 2 * hl];
                float esa = (float)pes[0], esb = (float)pes[1];
                float e1a = (float)pe1[0], e1b = (float)pe1[1];
                float e2a = (float)pe2[0], e2b = (float)pe2[1];
                res[t][0][r] = g[0]*e1a + g[1]*e1b + g[2]*esa + g[3]*esb + g[4]*e2a + g[5]*e2b;
                res[t][1][r] = g[6]*e1a + g[7]*e1b + g[8]*esa + g[9]*esb;
                res[t][2][r] = g[10]*e2a + g[11]*e2b + g[12]*esa + g[13]*esb;
            }
        }
        __syncthreads();   // all waves done reading old X

        // write new sh/t1/t2 in place
        #pragma unroll
        for (int t = 0; t < 2; ++t) {
            const int h0 = (wave * 2 + t) * 16;
            #pragma unroll
            for (int r = 0; r < 8; ++r) {
                int M = (lane < 16) ? r : r + 8;
                int h = h0 + (lane & 15);
                sX[0][M * XP + h] = (_Float16)res[t][0][r];
                sX[1][M * XP + h] = (_Float16)res[t][1][r];
                sX[2][M * XP + h] = (_Float16)res[t][2][r];
            }
        }
        __syncthreads();
    }

    // ---- heads: span(t1,5) | attr_start(t2,30) | attr_end(t2,30) ----
    for (int idx = tid; idx < 16 * 65; idx += 256) {
        int tok = idx / 65, oc = idx % 65;
        size_t gtok = (size_t)(m0 + tok);
        const _Float16* x; const float* W; float b; size_t off; int nc;
        if (oc < 5)       { x = &sX[1][tok * XP]; W = p.Wspan + oc;        nc = 5;  b = p.bspan[oc];     off = gtok * 5 + oc; }
        else if (oc < 35) { int c = oc - 5;  x = &sX[2][tok * XP]; W = p.Was + c; nc = 30; b = p.bas[c]; off = (size_t)TOKENS * 5  + gtok * 30 + c; }
        else              { int c = oc - 35; x = &sX[2][tok * XP]; W = p.Wae + c; nc = 30; b = p.bae[c]; off = (size_t)TOKENS * 35 + gtok * 30 + c; }
        float acc = b;
        for (int k = 0; k < H; ++k) acc += (float)x[k] * W[k * nc];
        p.out[off] = acc;
    }
}

extern "C" void kernel_launch(void* const* d_in, const int* in_sizes, int n_in,
                              void* d_out, int out_size, void* d_ws, size_t ws_size,
                              hipStream_t stream) {
    _Float16* wt = (_Float16*)d_ws;

    prep_weights<<<(2 * 3 * WTSZ + 255) / 256, 256, 0, stream>>>(
        (const float*)d_in[3], (const float*)d_in[5], (const float*)d_in[7], wt);

    PleArgs a;
    a.xsh  = (const float*)d_in[0];
    a.xt1  = (const float*)d_in[1];
    a.xt2  = (const float*)d_in[2];
    a.bsh  = (const float*)d_in[4];
    a.bt1  = (const float*)d_in[6];
    a.bt2  = (const float*)d_in[8];
    a.Wgsh = (const float*)d_in[9];   a.bgsh = (const float*)d_in[10];
    a.Wgt1 = (const float*)d_in[11];  a.bgt1 = (const float*)d_in[12];
    a.Wgt2 = (const float*)d_in[13];  a.bgt2 = (const float*)d_in[14];
    a.Wspan = (const float*)d_in[15]; a.bspan = (const float*)d_in[16];
    a.Was   = (const float*)d_in[17]; a.bas   = (const float*)d_in[18];
    a.Wae   = (const float*)d_in[19]; a.bae   = (const float*)d_in[20];
    a.wt  = wt;
    a.out = (float*)d_out;

    ple_fused<<<TOKENS / 16, 256, 0, stream>>>(a);
}
